// Pointnet_FP_60885456388434
// MI455X (gfx1250) — matrix operand order repaired and checked by reading, not compiled, not run on hardware
//
#include <hip/hip_runtime.h>
#include <hip/hip_bf16.h>

typedef __attribute__((ext_vector_type(16))) _Float16 v16h;
typedef __attribute__((ext_vector_type(8)))  _Float16 v8h;
typedef __attribute__((ext_vector_type(4)))  _Float16 h4;
typedef __attribute__((ext_vector_type(8)))  float    v8f;
typedef __attribute__((ext_vector_type(4)))  float    v4f;

#define BB  4
#define N1  16384
#define N2  4096
#define C1  64
#define C2  128
#define CIN 192      // C2 + C1
#define HID 128
#define OUTC 128

// ---------------------------------------------------------------------------
// Kernel 0: transpose weights to N-major f16: WT[n][k] = W[k][n]
// This makes WMMA B-fragments two contiguous 16B loads per lane.
// ---------------------------------------------------------------------------
__global__ __launch_bounds__(256) void wconv_kernel(
    const float* __restrict__ W1, const float* __restrict__ W2,
    _Float16* __restrict__ W1T, _Float16* __restrict__ W2T)
{
    int t = blockIdx.x * 256 + threadIdx.x;
    if (t < CIN * HID) {               // W1: [192,128] row-major
        int k = t / HID, n = t % HID;
        W1T[n * CIN + k] = (_Float16)W1[t];
    }
    if (t < HID * OUTC) {              // W2: [128,128] row-major
        int k = t / OUTC, n = t % OUTC;
        W2T[n * HID + k] = (_Float16)W2[t];
    }
}

// ---------------------------------------------------------------------------
// Kernel 1: 3-NN + inverse-distance interpolation + concat -> f16 rows [*,192]
// One thread per query point; xyz2 (48KB) staged in LDS; branchless top-3.
// ---------------------------------------------------------------------------
__global__ __launch_bounds__(256) void knn_interp_kernel(
    const float* __restrict__ xyz1, const float* __restrict__ xyz2,
    const float* __restrict__ points1, const float* __restrict__ points2,
    _Float16* __restrict__ A)
{
    __shared__ float xs[N2];
    __shared__ float ys[N2];
    __shared__ float zs[N2];

    const int b = blockIdx.y;
    const int q = blockIdx.x * 256 + threadIdx.x;  // query index in N1

    // stage xyz2 for this batch into LDS (coalesced flat read, scatter by comp)
    const float* p2 = xyz2 + (size_t)b * N2 * 3;
    for (int i = threadIdx.x; i < N2 * 3; i += 256) {
        float v = p2[i];
        int pt = i / 3, c = i - pt * 3;
        if (c == 0)      xs[pt] = v;
        else if (c == 1) ys[pt] = v;
        else             zs[pt] = v;
    }
    __syncthreads();

    const size_t row = (size_t)b * N1 + q;
    float qx = xyz1[row * 3 + 0];
    float qy = xyz1[row * 3 + 1];
    float qz = xyz1[row * 3 + 2];

    float d0 = 3.4e38f, d1 = 3.4e38f, d2 = 3.4e38f;
    int   i0 = 0, i1 = 0, i2 = 0;
    #pragma unroll 4
    for (int j = 0; j < N2; ++j) {
        float dx = qx - xs[j], dy = qy - ys[j], dz = qz - zs[j];
        float d = dx * dx + dy * dy + dz * dz;
        bool lt0 = d < d0, lt1 = d < d1, lt2 = d < d2;
        d2 = lt1 ? d1 : (lt2 ? d : d2);
        i2 = lt1 ? i1 : (lt2 ? j : i2);
        d1 = lt0 ? d0 : (lt1 ? d : d1);
        i1 = lt0 ? i0 : (lt1 ? j : i1);
        d0 = lt0 ? d  : d0;
        i0 = lt0 ? j  : i0;
    }

    d0 = fmaxf(d0, 1e-10f); d1 = fmaxf(d1, 1e-10f); d2 = fmaxf(d2, 1e-10f);
    float r0 = 1.0f / d0, r1 = 1.0f / d1, r2 = 1.0f / d2;
    float rn = 1.0f / (r0 + r1 + r2);
    float w0 = r0 * rn, w1 = r1 * rn, w2 = r2 * rn;

    // gather + interpolate C2 channels (rows are L2-resident), write f16
    const float* g0 = points2 + ((size_t)b * N2 + i0) * C2;
    const float* g1 = points2 + ((size_t)b * N2 + i1) * C2;
    const float* g2 = points2 + ((size_t)b * N2 + i2) * C2;
    _Float16* arow = A + row * CIN;
    #pragma unroll 4
    for (int c = 0; c < C2; c += 4) {
        v4f a0 = *(const v4f*)(g0 + c);
        v4f a1 = *(const v4f*)(g1 + c);
        v4f a2 = *(const v4f*)(g2 + c);
        h4 h;
        h.x = (_Float16)(w0 * a0.x + w1 * a1.x + w2 * a2.x);
        h.y = (_Float16)(w0 * a0.y + w1 * a1.y + w2 * a2.y);
        h.z = (_Float16)(w0 * a0.z + w1 * a1.z + w2 * a2.z);
        h.w = (_Float16)(w0 * a0.w + w1 * a1.w + w2 * a2.w);
        *(h4*)(arow + c) = h;
    }
    // concat points1 (C1 channels)
    const float* p1 = points1 + row * C1;
    #pragma unroll 4
    for (int c = 0; c < C1; c += 4) {
        v4f a = *(const v4f*)(p1 + c);
        h4 h;
        h.x = (_Float16)a.x; h.y = (_Float16)a.y;
        h.z = (_Float16)a.z; h.w = (_Float16)a.w;
        *(h4*)(arow + C2 + c) = h;
    }
}

// ---------------------------------------------------------------------------
// Kernel 2: fused MLP (192->128 ReLU -> 128->128 ReLU) on v_wmma_f32_16x16x32_f16
// 256 threads = 8 waves; wave = one 16-row strip; block = 128 rows x 128 cols.
// ---------------------------------------------------------------------------
__device__ __forceinline__ v16h load_frag(const _Float16* lo, const _Float16* hi)
{
    v8h a = *(const v8h*)lo;
    v8h b = *(const v8h*)hi;
    return __builtin_shufflevector(a, b, 0,1,2,3,4,5,6,7,8,9,10,11,12,13,14,15);
}

__global__ __launch_bounds__(256) void mlp_kernel(
    const _Float16* __restrict__ A, const _Float16* __restrict__ W1T,
    const _Float16* __restrict__ W2T,
    const float* __restrict__ b1, const float* __restrict__ b2,
    float* __restrict__ out)
{
    __shared__ _Float16 Hs[128 * HID];     // layer-1 activations (f16)  32KB
    __shared__ _Float16 W2s[OUTC * HID];   // W2^T staged in LDS         32KB

    const int lane  = threadIdx.x & 31;
    const int wave  = threadIdx.x >> 5;
    const int mrow  = lane & 15;           // row within 16x16 tile (A/B col for B)
    const int khalf = lane >> 4;           // 0: low-K half-wave, 1: high-K

    // stage W2^T into LDS, coalesced 16B chunks
    for (int i = threadIdx.x * 8; i < OUTC * HID; i += 256 * 8)
        *(v8h*)(W2s + i) = *(const v8h*)(W2T + i);
    __syncthreads();

    const size_t rowA = (size_t)blockIdx.x * 128 + wave * 16 + mrow;

    // ---------------- layer 1: [16x192] x [192x128] ----------------
    v8f acc[8] = {};
    for (int k0 = 0; k0 < CIN; k0 += 32) {
        // A fragment per ISA layout: lanes0-15 K=0..7 & 16..23, lanes16-31 K=8..15 & 24..31
        const _Float16* pa = A + rowA * CIN + k0 + khalf * 8;
        v16h af = load_frag(pa, pa + 16);
        #pragma unroll
        for (int n = 0; n < 8; ++n) {
            // B fragment: same K order as A (lanes0-15 K=0..7 & 16..23, lanes16-31 K=8..15 & 24..31) for column n*16+mrow
            const _Float16* pb = W1T + (size_t)(n * 16 + mrow) * CIN + k0 + khalf * 8;
            v16h bf = load_frag(pb, pb + 16);
            acc[n] = __builtin_amdgcn_wmma_f32_16x16x32_f16(
                         false, af, false, bf, (short)0, acc[n], false, false);
        }
    }

    // bias + ReLU, spill to LDS as f16 (C/D layout: VGPR r -> row r + khalf*8)
    #pragma unroll
    for (int n = 0; n < 8; ++n) {
        float bias = b1[n * 16 + mrow];
        #pragma unroll
        for (int r = 0; r < 8; ++r) {
            float v = fmaxf(acc[n][r] + bias, 0.0f);
            Hs[(wave * 16 + r + khalf * 8) * HID + n * 16 + mrow] = (_Float16)v;
        }
    }
    __syncthreads();

    // ---------------- layer 2: [16x128] x [128x128] ----------------
    v8f acc2[8] = {};
    for (int k0 = 0; k0 < HID; k0 += 32) {
        const _Float16* pa = Hs + (wave * 16 + mrow) * HID + k0 + khalf * 8;
        v16h af = load_frag(pa, pa + 16);
        #pragma unroll
        for (int n = 0; n < 8; ++n) {
            const _Float16* pb = W2s + (n * 16 + mrow) * HID + k0 + khalf * 8;
            v16h bf = load_frag(pb, pb + 16);
            acc2[n] = __builtin_amdgcn_wmma_f32_16x16x32_f16(
                          false, af, false, bf, (short)0, acc2[n], false, false);
        }
    }

    // bias + ReLU, store f32 result
    const size_t obase = (size_t)blockIdx.x * 128 + wave * 16;
    #pragma unroll
    for (int n = 0; n < 8; ++n) {
        float bias = b2[n * 16 + mrow];
        #pragma unroll
        for (int r = 0; r < 8; ++r) {
            out[(obase + r + khalf * 8) * OUTC + n * 16 + mrow] =
                fmaxf(acc2[n][r] + bias, 0.0f);
        }
    }
}

// ---------------------------------------------------------------------------
extern "C" void kernel_launch(void* const* d_in, const int* in_sizes, int n_in,
                              void* d_out, int out_size, void* d_ws, size_t ws_size,
                              hipStream_t stream)
{
    const float* xyz1    = (const float*)d_in[0];
    const float* xyz2    = (const float*)d_in[1];
    const float* points1 = (const float*)d_in[2];
    const float* points2 = (const float*)d_in[3];
    const float* W1      = (const float*)d_in[4];
    const float* b1      = (const float*)d_in[5];
    const float* W2      = (const float*)d_in[6];
    const float* b2      = (const float*)d_in[7];
    float* out = (float*)d_out;

    // workspace: A f16 [B*N1, 192], W1T f16 [128,192], W2T f16 [128,128]
    _Float16* A   = (_Float16*)d_ws;
    _Float16* W1T = A + (size_t)BB * N1 * CIN;
    _Float16* W2T = W1T + (size_t)HID * CIN;

    wconv_kernel<<<(CIN * HID + 255) / 256, 256, 0, stream>>>(W1, W2, W1T, W2T);

    dim3 gknn(N1 / 256, BB);
    knn_interp_kernel<<<gknn, 256, 0, stream>>>(xyz1, xyz2, points1, points2, A);

    mlp_kernel<<<(BB * N1) / 128, 256, 0, stream>>>(A, W1T, W2T, b1, b2, out);
}
